// TrajDiscriminator_5446018531386
// MI455X (gfx1250) — matrix operand order repaired and checked by reading, not compiled, not run on hardware
//
#include <hip/hip_runtime.h>
#include <hip/hip_bf16.h>

// GRU trajectory discriminator, fully fused for MI455X (gfx1250, wave32).
// One wave = 16 batch rows. Per time step: 48 v_wmma_f32_16x16x32_f16.
//   r,z : [inp|h](16x128) @ Wc cols 0..127  -> 32 WMMA (K fused via combined weights)
//   n   : gh_n (K=64 over h) -> *r element-wise -> +i_n via WMMA C operand -> 16 WMMA
// Weights + biases live in LDS as f16/f32. h lives in registers (f32, C/D layout)
// + an f16 LDS mirror re-read in A-fragment layout. B-fragment loads grouped 4-wide.

#define TT 24
#define B_TOT 65536
#define WAVES_PER_BLK 2
#define ROWS_PER_WAVE 16
#define WC_STRIDE 136   // f16 elems/row (192 rows); multiple of 8 -> 16B aligned rows
#define SH_STRIDE 136   // f16 elems/row of per-wave [inp(64)|h(64)] staging

typedef __attribute__((ext_vector_type(16))) _Float16 v16h;
typedef __attribute__((ext_vector_type(8)))  _Float16 v8h;
typedef __attribute__((ext_vector_type(8)))  float    v8f;

union frag32 { v16h v; v8h h[2]; };

__device__ __forceinline__ float sigmoidf_(float x) { return 1.0f / (1.0f + __expf(-x)); }
__device__ __forceinline__ float tanhf_(float x) {
  float e = __expf(2.0f * x);              // tanh = 1 - 2/(e^{2x}+1); saturates cleanly
  return 1.0f - 2.0f / (e + 1.0f);
}
__device__ __forceinline__ float leakyf_(float x) { return x > 0.0f ? x : 0.2f * x; }

// B fragment (16-bit, 32x16): lane col = ln%16; lanes 0-15 hold K=0..15,
// lanes 16-31 hold K=16..31, packed 2/VGPR -> two contiguous 16B LDS reads.
__device__ __forceinline__ v16h loadB(const _Float16* Wc, int row, int kt, int hi) {
  const _Float16* p = Wc + row * WC_STRIDE + kt * 32 + hi * 16;
  frag32 bf;
  bf.h[0] = *(const v8h*)(p);
  bf.h[1] = *(const v8h*)(p + 8);
  return bf.v;
}

// A fragment (16-bit, 16x32): lane row m = ln%16; halves 0..15 -> K = 16*hi + c
// (matches loadB): two contiguous 16B reads at koff, koff+8.
__device__ __forceinline__ v16h loadA(const _Float16* S, int nn, int kt, int hi) {
  const _Float16* p = S + nn * SH_STRIDE + kt * 32 + hi * 16;
  frag32 af;
  af.h[0] = *(const v8h*)(p);
  af.h[1] = *(const v8h*)(p + 8);
  return af.v;
}

extern "C" __global__ __launch_bounds__(WAVES_PER_BLK * 32)
__attribute__((amdgpu_num_vgpr(256)))   // stay in the low bank: no s_set_vgpr_msb
void traj_gru_fused(const float* __restrict__ x,          // [B,T,2]
                    const float* __restrict__ W_loc,      // [2,32]
                    const float* __restrict__ b_loc,      // [32]
                    const float* __restrict__ time_table, // [24,32]
                    const float* __restrict__ W_ih,       // [192,64]
                    const float* __restrict__ W_hh,       // [192,64]
                    const float* __restrict__ b_ih,       // [192]
                    const float* __restrict__ b_hh,       // [192]
                    const float* __restrict__ W_pred,     // [64,1]
                    const float* __restrict__ b_pred,     // [1]
                    float* __restrict__ out)              // [B]
{
  // Combined weights, transposed: Wc[n][k], n=0..191 gate col, k: 0..63 = W_ih[n][k],
  // 64..127 = W_hh[n][k-64]. Row stride padded (136) to spread LDS banks.
  __shared__ __align__(16) _Float16 Wc[192 * WC_STRIDE];
  __shared__ __align__(16) _Float16 shs[WAVES_PER_BLK][16 * SH_STRIDE];
  __shared__ __align__(16) _Float16 ltab[TT * 32];   // leaky(time_table)
  __shared__ float wloc[96];                         // W_loc row0 | row1 | b_loc
  // bias table: [0..127]  = b_ih+b_hh for gate cols 0..127 (r,z)
  //             [128..191]= b_hh[128..191] (gh_n seed), [192..255]= b_ih[128..191] (i_n seed)
  __shared__ float btab[256];

  const int tid = threadIdx.x;
  const int nthr = WAVES_PER_BLK * 32;

  for (int i = tid; i < 192 * 128; i += nthr) {
    int n = i >> 7, k = i & 127;
    float v = (k < 64) ? W_ih[n * 64 + k] : W_hh[n * 64 + (k - 64)];
    Wc[n * WC_STRIDE + k] = (_Float16)v;
  }
  for (int i = tid; i < TT * 32; i += nthr) ltab[i] = (_Float16)leakyf_(time_table[i]);
  for (int i = tid; i < 96; i += nthr) wloc[i] = (i < 64) ? W_loc[i] : b_loc[i - 64];
  for (int i = tid; i < 256; i += nthr) {
    float v;
    if (i < 128)      v = b_ih[i] + b_hh[i];
    else if (i < 192) v = b_hh[i];          // cols 128..191
    else              v = b_ih[i - 64];     // cols 128..191
    btab[i] = v;
  }
  __syncthreads();

  const int wave = tid >> 5;
  const int ln   = tid & 31;
  const int nn   = ln & 15;   // lane within 16-group (== N col / A row M)
  const int hi   = ln >> 4;   // which 16-lane half
  _Float16* S = shs[wave];
  const int b0 = (blockIdx.x * WAVES_PER_BLK + wave) * ROWS_PER_WAVE;

  // h = 0 : registers (C/D layout, 4 N-tiles) + f16 mirror in LDS cols 64..127
  v8f hreg[4];
#pragma unroll
  for (int j = 0; j < 4; ++j)
#pragma unroll
    for (int e = 0; e < 8; ++e) hreg[j][e] = 0.0f;
  for (int c = 0; c < 32; ++c) S[nn * SH_STRIDE + 64 + hi * 32 + c] = (_Float16)0.0f;

  const float* xrow = x + (size_t)(b0 + nn) * (TT * 2);

#pragma unroll 1
  for (int t = 0; t < TT; ++t) {
    // ---- stage inp = [leaky(x @ W_loc + b_loc) | leaky(time_table[t])] as f16
    // lane covers row nn, cols 16*hi .. 16*hi+15 of each 32-wide half.
    float x0 = xrow[2 * t + 0], x1 = xrow[2 * t + 1];
    if (t + 1 < TT) __builtin_prefetch(&xrow[2 * (t + 1)], 0, 3);  // global_prefetch_b8
#pragma unroll
    for (int c = 0; c < 16; ++c) {
      int k = hi * 16 + c;
      float e = leakyf_(x0 * wloc[k] + x1 * wloc[32 + k] + wloc[64 + k]);
      S[nn * SH_STRIDE + k]      = (_Float16)e;
      S[nn * SH_STRIDE + 32 + k] = ltab[t * 32 + k];
    }
    asm volatile("s_wait_dscnt 0" ::: "memory");  // cross-lane LDS visibility

    v8f acc[12];

    // ==== Phase 1: h-dependent K-tiles (kt=2,3).
    frag32 af2, af3;
    af2.v = loadA(S, nn, 2, hi);
    af3.v = loadA(S, nn, 3, hi);

    // r,z over h K-tiles
#pragma unroll
    for (int j = 0; j < 8; ++j) {
      float b = btab[16 * j + nn];
#pragma unroll
      for (int e = 0; e < 8; ++e) acc[j][e] = b;
    }
#pragma unroll
    for (int kt = 2; kt < 4; ++kt) {
      const v16h a = (kt == 2) ? af2.v : af3.v;
      v16h bf[4];
#pragma unroll
      for (int j = 0; j < 4; ++j) bf[j] = loadB(Wc, 16 * j + nn, kt, hi);
#pragma unroll
      for (int j = 0; j < 4; ++j)
        acc[j] = __builtin_amdgcn_wmma_f32_16x16x32_f16(
            false, a, false, bf[j], (short)0, acc[j], false, false);
#pragma unroll
      for (int j = 0; j < 4; ++j) bf[j] = loadB(Wc, 64 + 16 * j + nn, kt, hi);
#pragma unroll
      for (int j = 0; j < 4; ++j)
        acc[4 + j] = __builtin_amdgcn_wmma_f32_16x16x32_f16(
            false, a, false, bf[j], (short)0, acc[4 + j], false, false);
    }
    // gh_n (+b_hh_n) over h K-tiles; init right before first use
#pragma unroll
    for (int j = 0; j < 4; ++j) {
      float b = btab[128 + 16 * j + nn];
#pragma unroll
      for (int e = 0; e < 8; ++e) acc[8 + j][e] = b;
    }
#pragma unroll
    for (int kt = 2; kt < 4; ++kt) {
      const v16h a = (kt == 2) ? af2.v : af3.v;
      v16h bf[4];
#pragma unroll
      for (int j = 0; j < 4; ++j) bf[j] = loadB(Wc, 128 + 16 * j + nn, kt, hi);
#pragma unroll
      for (int j = 0; j < 4; ++j)
        acc[8 + j] = __builtin_amdgcn_wmma_f32_16x16x32_f16(
            false, a, false, bf[j], (short)0, acc[8 + j], false, false);
    }

    // ==== Phase 2: inp K-tiles (kt=0,1): finish r,z.  (af2/af3 are dead now)
    frag32 af0, af1;
    af0.v = loadA(S, nn, 0, hi);
    af1.v = loadA(S, nn, 1, hi);
#pragma unroll
    for (int kt = 0; kt < 2; ++kt) {
      const v16h a = (kt == 0) ? af0.v : af1.v;
      v16h bf[4];
#pragma unroll
      for (int j = 0; j < 4; ++j) bf[j] = loadB(Wc, 16 * j + nn, kt, hi);
#pragma unroll
      for (int j = 0; j < 4; ++j)
        acc[j] = __builtin_amdgcn_wmma_f32_16x16x32_f16(
            false, a, false, bf[j], (short)0, acc[j], false, false);
#pragma unroll
      for (int j = 0; j < 4; ++j) bf[j] = loadB(Wc, 64 + 16 * j + nn, kt, hi);
#pragma unroll
      for (int j = 0; j < 4; ++j)
        acc[4 + j] = __builtin_amdgcn_wmma_f32_16x16x32_f16(
            false, a, false, bf[j], (short)0, acc[4 + j], false, false);
    }

    // ---- gates r (tiles 0..3), z (tiles 4..7)
#pragma unroll
    for (int j = 0; j < 8; ++j)
#pragma unroll
      for (int e = 0; e < 8; ++e) acc[j][e] = sigmoidf_(acc[j][e]);

    // ---- n pre-activation seed: r * (gh_n + b_hh_n) + b_ih_n
#pragma unroll
    for (int j = 0; j < 4; ++j) {
      float b = btab[192 + 16 * j + nn];
#pragma unroll
      for (int e = 0; e < 8; ++e)
        acc[8 + j][e] = acc[j][e] * acc[8 + j][e] + b;
    }

    // ==== Phase 3: += i_n over inp K-tiles (kt=0,1), via WMMA C operand.
#pragma unroll
    for (int kt = 0; kt < 2; ++kt) {
      const v16h a = (kt == 0) ? af0.v : af1.v;
      v16h bf[4];
#pragma unroll
      for (int j = 0; j < 4; ++j) bf[j] = loadB(Wc, 128 + 16 * j + nn, kt, hi);
#pragma unroll
      for (int j = 0; j < 4; ++j)
        acc[8 + j] = __builtin_amdgcn_wmma_f32_16x16x32_f16(
            false, a, false, bf[j], (short)0, acc[8 + j], false, false);
    }

    // ---- h' = (1-z)*tanh(n) + z*h ; refresh f16 mirror in LDS (C/D -> row-major)
#pragma unroll
    for (int j = 0; j < 4; ++j)
#pragma unroll
      for (int e = 0; e < 8; ++e) {
        float n_ = tanhf_(acc[8 + j][e]);
        float z_ = acc[4 + j][e];
        float hn = (1.0f - z_) * n_ + z_ * hreg[j][e];
        hreg[j][e] = hn;
        // C/D layout: element (M = e + 8*hi, N = nn) of N-tile j -> h col 16*j+nn
        S[(e + 8 * hi) * SH_STRIDE + 64 + 16 * j + nn] = (_Float16)hn;
      }
    asm volatile("s_wait_dscnt 0" ::: "memory");
  }

  // ---- head: out[b0+m] = sigmoid(sum_n h[m,n] * W_pred[n] + b_pred)
  float wp[4];
#pragma unroll
  for (int j = 0; j < 4; ++j) wp[j] = W_pred[16 * j + nn];
  float p[8];
#pragma unroll
  for (int e = 0; e < 8; ++e)
    p[e] = hreg[0][e] * wp[0] + hreg[1][e] * wp[1] + hreg[2][e] * wp[2] + hreg[3][e] * wp[3];
  // butterfly reduce across the 16 lanes of each half (wave32 shuffles)
#pragma unroll
  for (int mask = 1; mask < 16; mask <<= 1)
#pragma unroll
    for (int e = 0; e < 8; ++e) p[e] += __shfl_xor(p[e], mask, 32);
  if (nn == 0) {
    float bp = b_pred[0];
#pragma unroll
    for (int e = 0; e < 8; ++e)
      out[b0 + 8 * hi + e] = sigmoidf_(p[e] + bp);
  }
}

extern "C" void kernel_launch(void* const* d_in, const int* in_sizes, int n_in,
                              void* d_out, int out_size, void* d_ws, size_t ws_size,
                              hipStream_t stream) {
  (void)in_sizes; (void)n_in; (void)d_ws; (void)ws_size; (void)out_size;
  const float* x          = (const float*)d_in[0];
  const float* W_loc      = (const float*)d_in[1];
  const float* b_loc      = (const float*)d_in[2];
  const float* time_table = (const float*)d_in[3];
  const float* W_ih       = (const float*)d_in[4];
  const float* W_hh       = (const float*)d_in[5];
  const float* b_ih       = (const float*)d_in[6];
  const float* b_hh       = (const float*)d_in[7];
  const float* W_pred     = (const float*)d_in[8];
  const float* b_pred     = (const float*)d_in[9];

  const int rows_per_block = WAVES_PER_BLK * ROWS_PER_WAVE;  // 32
  dim3 grid(B_TOT / rows_per_block);                         // 2048
  dim3 block(WAVES_PER_BLK * 32);                            // 64 (2 waves)
  traj_gru_fused<<<grid, block, 0, stream>>>(
      x, W_loc, b_loc, time_table, W_ih, W_hh, b_ih, b_hh, W_pred, b_pred,
      (float*)d_out);
}
